// PredictionBlockPS_89550068122291
// MI455X (gfx1250) — matrix operand order repaired and checked by reading, not compiled, not run on hardware
//
#include <hip/hip_runtime.h>
#include <hip/hip_bf16.h>
#include <math.h>
#include <stdint.h>

typedef __attribute__((ext_vector_type(16))) _Float16 v16h;
typedef __attribute__((ext_vector_type(8)))  _Float16 v8h;
typedef __attribute__((ext_vector_type(8)))  float    v8f;

// ---------------- problem constants ----------------
constexpr int B_    = 8;
constexpr int C_    = 576;
constexpr int CHLF  = 288;
constexpr int H_    = 32;
constexpr int W_    = 32;
constexpr int NJ_   = 16;
constexpr int NHEAT = 48;    // NJ*(1+NC)
constexpr int CHEAD = 256;   // ND*NJ
constexpr int NBLK  = 8;
constexpr float ALPHA_ = 0.8f;

// ---------------- WMMA GEMM ------------------------------------------------
// out[o][p] = bn(W[o][:] . f(in[:,p])) (+ residual adds)
// f16 inputs, f32 accumulate. Block = 128 threads = 4 waves (2x2), tile 64x64.
// LDS tiles are stored pre-permuted into WMMA fragment layout: per (subtile,
// lane) the 16 halfs of a fragment are contiguous (32B) -> 2x ds_load_b128.
// A tile (already f16 in memory) is staged with async global->LDS copies.
constexpr int TILE  = 64;
constexpr int KSTEP = 32;

__global__ __launch_bounds__(128) void gemm_pw_kernel(
    const float* __restrict__ in1, const float* __restrict__ in2,
    const _Float16* __restrict__ Wt,        // rows padded to multiple of 64, zero-filled
    const float* __restrict__ gamma, const float* __restrict__ beta,
    const float* __restrict__ add1, const float* __restrict__ add2,
    float* __restrict__ out, int Cin, int Cout, int P, int relu_in)
{
  // 4 m-subtiles x 32 lanes x 16 halfs  (4KB each)
  __shared__ __attribute__((aligned(32))) _Float16 laA[4 * 32 * 16];
  __shared__ __attribute__((aligned(32))) _Float16 laB[4 * 32 * 16];

  const int bb = blockIdx.z;
  in1 += (size_t)bb * Cin * P;
  if (in2)  in2  += (size_t)bb * Cin * P;
  if (add1) add1 += (size_t)bb * Cout * P;
  if (add2) add2 += (size_t)bb * Cout * P;
  out += (size_t)bb * Cout * P;

  const int mBase = blockIdx.y * TILE;
  const int pBase = blockIdx.x * TILE;
  const int tid   = threadIdx.x;
  const int lane  = tid & 31;
  const int wave  = tid >> 5;
  const int wy    = wave >> 1, wx = wave & 1;
  const int mr    = lane & 15;
  const bool hiHalf = lane >= 16;

  // precompute per-thread A staging addresses (2 chunks: c = tid, tid+128)
  // chunk c: m = c>>2, k-group g = c&3 ; 8 halfs land on one lane, slots s0..s0+7
  uint32_t aLds[2];
  int aM[2], aK0[2];
#pragma unroll
  for (int i = 0; i < 2; ++i) {
    int c  = tid + i * 128;
    int m  = c >> 2;
    int g  = c & 3;
    int dlane = (m & 15) + ((g & 1) << 4);
    int s0    = (g >> 1) << 3;
    int mt    = m >> 4;
    aLds[i] = (uint32_t)(uintptr_t)&laA[((mt * 32 + dlane) << 4) + s0];
    aM[i]  = m;
    aK0[i] = g * 8;
  }

  v8f zero = {0.f,0.f,0.f,0.f,0.f,0.f,0.f,0.f};
  v8f acc[2][2] = {{zero, zero}, {zero, zero}};

  for (int kk = 0; kk < Cin; kk += KSTEP) {
    // ---- stage A: async global->LDS, 16B per lane per chunk ----
#pragma unroll
    for (int i = 0; i < 2; ++i) {
      uint64_t ga = (uint64_t)(uintptr_t)(Wt + (size_t)(mBase + aM[i]) * Cin + kk + aK0[i]);
      asm volatile("global_load_async_to_lds_b128 %0, %1, off"
                   :: "v"(aLds[i]), "v"(ga) : "memory");
    }
    // ---- stage B: 32 k x 64 p of activations, fp32->f16 fused add/relu ----
    // chunk c: p = c&63, k0 = (c>>6)*8 ; 8 consecutive k -> one lane, slots contiguous
    for (int c = tid; c < 256; c += 128) {
      int p  = c & 63;
      int k0 = (c >> 6) << 3;
      v8h hv;
#pragma unroll
      for (int i = 0; i < 8; ++i) {
        size_t off = (size_t)(kk + k0 + i) * P + pBase + p;
        float v = in1[off];
        if (in2) v += in2[off];
        if (relu_in) v = fmaxf(v, 0.f);
        hv[i] = (_Float16)v;
      }
      int dlane = (p & 15) + (((k0 >> 3) & 1) << 4);
      int s0    = (k0 >> 4) << 3;
      int nt    = p >> 4;
      *(v8h*)(&laB[((nt * 32 + dlane) << 4) + s0]) = hv;
    }
    if (kk + KSTEP < Cin)   // hint next activation tile toward the caches
      __builtin_prefetch(in1 + (size_t)(kk + KSTEP + (tid >> 6)) * P + pBase + (tid & 63), 0, 1);
    asm volatile("s_wait_asynccnt 0x0" ::: "memory");
    __syncthreads();

    // ---- fragment loads: contiguous 32B per lane -> 2x ds_load_b128 ------
    v16h afr[2], bfr[2];
#pragma unroll
    for (int mt = 0; mt < 2; ++mt)
      afr[mt] = *(const v16h*)(&laA[(((wy * 2 + mt) * 32 + lane) << 4)]);
#pragma unroll
    for (int nt = 0; nt < 2; ++nt)
      bfr[nt] = *(const v16h*)(&laB[(((wx * 2 + nt) * 32 + lane) << 4)]);

#pragma unroll
    for (int mt = 0; mt < 2; ++mt)
#pragma unroll
      for (int nt = 0; nt < 2; ++nt)
        acc[mt][nt] = __builtin_amdgcn_wmma_f32_16x16x32_f16(
            false, afr[mt], false, bfr[nt], (short)0, acc[mt][nt], false, false);
    __syncthreads();
  }

  // epilogue: bn + residual adds (C/D layout: vgpr v -> M=v(+8 for hi lanes))
#pragma unroll
  for (int mt = 0; mt < 2; ++mt) {
#pragma unroll
    for (int nt = 0; nt < 2; ++nt) {
      int pG = pBase + wx * 32 + nt * 16 + mr;
#pragma unroll
      for (int v = 0; v < 8; ++v) {
        int mG = mBase + wy * 32 + mt * 16 + v + (hiHalf ? 8 : 0);
        if (mG < Cout) {
          float val = acc[mt][nt][v];
          if (gamma) val = val * gamma[mG] + beta[mG];
          size_t off = (size_t)mG * P + pG;
          if (add1) val += add1[off];
          if (add2) val += add2[off];
          out[off] = val;
        }
      }
    }
  }
}

// ---------------- depthwise 5x5, pad 2, fused relu(in1 [+ in2]) -------------
__global__ __launch_bounds__(256) void dw5x5_kernel(
    const float* __restrict__ in1, const float* __restrict__ in2,
    const float* __restrict__ w, float* __restrict__ out,
    int C, int H, int W, int total)
{
  int idx = blockIdx.x * 256 + threadIdx.x;
  if (idx >= total) return;
  int wp = idx % W;
  int hp = (idx / W) % H;
  int c  = (idx / (W * H)) % C;
  int b  = idx / (W * H * C);
  const float* ip1 = in1 + ((size_t)b * C + c) * H * W;
  const float* ip2 = in2 ? in2 + ((size_t)b * C + c) * H * W : nullptr;
  const float* wc = w + c * 25;
  float acc = 0.f;
#pragma unroll
  for (int ky = 0; ky < 5; ++ky) {
    int ih = hp + ky - 2;
    if (ih < 0 || ih >= H) continue;
#pragma unroll
    for (int kx = 0; kx < 5; ++kx) {
      int iw = wp + kx - 2;
      if (iw < 0 || iw >= W) continue;
      float v = ip1[ih * W + iw];
      if (ip2) v += ip2[ih * W + iw];
      v = fmaxf(v, 0.f);
      acc += v * wc[ky * 5 + kx];
    }
  }
  out[idx] = acc;
}

// ---------------- maxpool 2x2 stride 2 --------------------------------------
__global__ __launch_bounds__(256) void maxpool2_kernel(
    const float* __restrict__ in, float* __restrict__ out, int C, int H, int W, int total)
{
  int idx = blockIdx.x * 256 + threadIdx.x;
  if (idx >= total) return;
  int Wo = W >> 1, Ho = H >> 1;
  int wo = idx % Wo;
  int ho = (idx / Wo) % Ho;
  int c  = (idx / (Wo * Ho)) % C;
  int b  = idx / (Wo * Ho * C);
  const float* ip = in + ((size_t)b * C + c) * H * W;
  int h2 = ho * 2, w2 = wo * 2;
  float m = fmaxf(fmaxf(ip[h2 * W + w2],       ip[h2 * W + w2 + 1]),
                  fmaxf(ip[(h2 + 1) * W + w2], ip[(h2 + 1) * W + w2 + 1]));
  out[idx] = m;
}

// ---------------- 2x nearest upsample ---------------------------------------
__global__ __launch_bounds__(256) void upsample2_kernel(
    const float* __restrict__ in, float* __restrict__ out, int C, int Hin, int Win, int total)
{
  int idx = blockIdx.x * 256 + threadIdx.x;
  if (idx >= total) return;
  int Wo = Win * 2, Ho = Hin * 2;
  int wo = idx % Wo;
  int ho = (idx / Wo) % Ho;
  int c  = (idx / (Wo * Ho)) % C;
  int b  = idx / (Wo * Ho * C);
  out[idx] = in[(((size_t)b * C + c) * Hin + (ho >> 1)) * Win + (wo >> 1)];
}

// ---------------- elementwise add -------------------------------------------
__global__ __launch_bounds__(256) void add_kernel(
    const float* __restrict__ a, const float* __restrict__ b, float* __restrict__ o, int n)
{
  int i = blockIdx.x * 256 + threadIdx.x;
  if (i < n) o[i] = a[i] + b[i];
}

// ---------------- fp32 -> f16 weight convert with row padding ---------------
__global__ __launch_bounds__(256) void cvt_pad_kernel(
    const float* __restrict__ in, _Float16* __restrict__ out,
    int Cout, int Cin, int CoutPad)
{
  int i = blockIdx.x * 256 + threadIdx.x;
  if (i >= CoutPad * Cin) return;
  int r = i / Cin;
  out[i] = (r < Cout) ? (_Float16)in[i] : (_Float16)0.f;
}

// ---------------- softargmax stats per (batch, heat channel) ----------------
// stats[b*48+ch] = {sum e, sum e*x, sum e*y, max}
__global__ __launch_bounds__(256) void head_stats_kernel(
    const float* __restrict__ Hbuf, float* __restrict__ stats)
{
  __shared__ float r0[256], r1[256], r2[256];
  int b  = blockIdx.x / NHEAT;
  int ch = blockIdx.x % NHEAT;
  const float* hp = Hbuf + ((size_t)b * CHEAD + ch) * (H_ * W_);
  int t = threadIdx.x;
  float v[4];
  float mx = -3.4e38f;
#pragma unroll
  for (int i = 0; i < 4; ++i) { v[i] = hp[t + i * 256]; mx = fmaxf(mx, v[i]); }
  r0[t] = mx; __syncthreads();
  for (int s = 128; s > 0; s >>= 1) { if (t < s) r0[t] = fmaxf(r0[t], r0[t + s]); __syncthreads(); }
  float smax = r0[0]; __syncthreads();
  float se = 0.f, sex = 0.f, sey = 0.f;
#pragma unroll
  for (int i = 0; i < 4; ++i) {
    int p = t + i * 256;
    float e = expf(v[i] - smax);
    se  += e;
    sex += e * (float)(p & 31) * (1.f / 31.f);
    sey += e * (float)(p >> 5) * (1.f / 31.f);
  }
  r0[t] = se; r1[t] = sex; r2[t] = sey; __syncthreads();
  for (int s = 128; s > 0; s >>= 1) {
    if (t < s) { r0[t] += r0[t + s]; r1[t] += r1[t + s]; r2[t] += r2[t + s]; }
    __syncthreads();
  }
  if (t == 0) {
    float* sp = stats + (size_t)blockIdx.x * 4;
    sp[0] = r0[0]; sp[1] = r1[0]; sp[2] = r2[0]; sp[3] = smax;
  }
}

// ---------------- combine heads -> d_out ------------------------------------
__global__ __launch_bounds__(128) void head_out_kernel(
    const float* __restrict__ stats, float* __restrict__ out, int blk)
{
  int t = threadIdx.x;
  if (t >= B_ * NJ_) return;
  int b = t >> 4, j = t & 15;
  const float* s0 = stats + ((size_t)b * NHEAT + j) * 4;
  const float* s1 = stats + ((size_t)b * NHEAT + NJ_ + j) * 4;
  const float* s2 = stats + ((size_t)b * NHEAT + 2 * NJ_ + j) * 4;
  float px = s0[1] / s0[0], py = s0[2] / s0[0];
  float vs = 1.f / (1.f + expf(-s0[3]));
  float p0x = s1[1] / s1[0], p0y = s1[2] / s1[0], v0 = 1.f / (1.f + expf(-s1[3]));
  float p1x = s2[1] / s2[0], p1y = s2[2] / s2[0], v1 = 1.f / (1.f + expf(-s2[3]));
  float den = v0 + v1 + 1e-6f;
  float cx = (p0x * v0 + p1x * v1) / den;
  float cy = (p0y * v0 + p1y * v1) / den;
  float* op = out + (((size_t)blk * B_ + b) * NJ_ + j) * 3;
  op[0] = ALPHA_ * px + (1.f - ALPHA_) * cx;
  op[1] = ALPHA_ * py + (1.f - ALPHA_) * cy;
  op[2] = vs;
}

// ============================================================================
extern "C" void kernel_launch(void* const* d_in, const int* in_sizes, int n_in,
                              void* d_out, int out_size, void* d_ws, size_t ws_size,
                              hipStream_t stream) {
  (void)in_sizes; (void)n_in; (void)out_size; (void)ws_size;
  // -------- inputs in setup_inputs() dict (insertion) order, depth-first ----
  const float* x_in   = (const float*)d_in[0];
  const float* a_dw   = (const float*)d_in[1];
  const float* a_pw   = (const float*)d_in[2];
  const float* a_g    = (const float*)d_in[3];
  const float* a_b    = (const float*)d_in[4];
  const float* low_w  = (const float*)d_in[5];
  const float* low_g  = (const float*)d_in[6];
  const float* low_b  = (const float*)d_in[7];
  const float* b1_dw  = (const float*)d_in[8];
  const float* b1_pw  = (const float*)d_in[9];
  const float* b1_g   = (const float*)d_in[10];
  const float* b1_b   = (const float*)d_in[11];
  const float* c1_dw  = (const float*)d_in[12];
  const float* c1_pw  = (const float*)d_in[13];
  const float* c1_g   = (const float*)d_in[14];
  const float* c1_b   = (const float*)d_in[15];
  const float* c2_dw  = (const float*)d_in[16];
  const float* c2_pw  = (const float*)d_in[17];
  const float* c2_g   = (const float*)d_in[18];
  const float* c2_b   = (const float*)d_in[19];
  const float* bu_dw  = (const float*)d_in[20];
  const float* bu_pw  = (const float*)d_in[21];
  const float* bu_g   = (const float*)d_in[22];
  const float* bu_b   = (const float*)d_in[23];
  const float* sc_w   = (const float*)d_in[24];
  const float* sc_g   = (const float*)d_in[25];
  const float* sc_b   = (const float*)d_in[26];
  const float* l10_dw = (const float*)d_in[27];
  const float* l10_pw = (const float*)d_in[28];
  const float* l10_g  = (const float*)d_in[29];
  const float* l10_b  = (const float*)d_in[30];
  const float* l11_w  = (const float*)d_in[31];
  const float* l12_w  = (const float*)d_in[32];
  const float* l12_g  = (const float*)d_in[33];
  const float* l12_b  = (const float*)d_in[34];
  float* outp = (float*)d_out;

  // -------- workspace carve-up ---------------------------------------------
  const size_t FULL_E = (size_t)B_ * C_ * H_ * W_;          // 4,718,592
  const size_t HB_E   = (size_t)B_ * CHEAD * H_ * W_;       // 2,097,152
  const size_t SB_E   = (size_t)B_ * C_ * 16 * 16;          // 1,179,648
  const size_t LB_E   = (size_t)B_ * CHLF * 16 * 16;        //   589,824
  const size_t MQ_E   = (size_t)B_ * CHLF * 8 * 8;          //   147,456
  char* cur = (char*)d_ws;
  auto carve = [&](size_t bytes) { char* p = cur; cur += (bytes + 255) & ~(size_t)255; return p; };
  float* Xc   = (float*)carve(FULL_E * 4);
  float* A1   = (float*)carve(FULL_E * 4);
  float* A2   = (float*)carve(FULL_E * 4);
  float* Hbuf = (float*)carve(HB_E * 4);
  float* S1   = (float*)carve(SB_E * 4);
  float* S2   = (float*)carve(SB_E * 4);
  float* L1   = (float*)carve(LB_E * 4);
  float* L2   = (float*)carve(LB_E * 4);
  float* L3   = (float*)carve(LB_E * 4);
  float* M1   = (float*)carve(MQ_E * 4);
  float* M2   = (float*)carve(MQ_E * 4);
  float* M3   = (float*)carve(MQ_E * 4);
  float* stats= (float*)carve((size_t)B_ * NHEAT * 4 * 4);
  // f16 weights, Cout padded to multiple of 64 rows (zero-filled)
  _Float16* Wa   = (_Float16*)carve((size_t)576 * 576 * 2);   // 576 -> 576
  _Float16* Wlow = (_Float16*)carve((size_t)320 * 576 * 2);   // 288 -> 320
  _Float16* Wb1  = (_Float16*)carve((size_t)320 * 288 * 2);
  _Float16* Wc1  = (_Float16*)carve((size_t)320 * 288 * 2);
  _Float16* Wc2  = (_Float16*)carve((size_t)320 * 288 * 2);
  _Float16* Wbu  = (_Float16*)carve((size_t)576 * 288 * 2);
  _Float16* Wsc  = (_Float16*)carve((size_t)576 * 288 * 2);
  _Float16* Wl10 = (_Float16*)carve((size_t)576 * 576 * 2);
  _Float16* Wl11 = (_Float16*)carve((size_t)256 * 576 * 2);
  _Float16* Wl12 = (_Float16*)carve((size_t)576 * 256 * 2);

  auto cvt = [&](const float* s, _Float16* d, int Cout, int Cin, int CoutPad) {
    int n = CoutPad * Cin;
    cvt_pad_kernel<<<(n + 255) / 256, 256, 0, stream>>>(s, d, Cout, Cin, CoutPad);
  };
  cvt(a_pw,   Wa,   576, 576, 576);
  cvt(low_w,  Wlow, 288, 576, 320);
  cvt(b1_pw,  Wb1,  288, 288, 320);
  cvt(c1_pw,  Wc1,  288, 288, 320);
  cvt(c2_pw,  Wc2,  288, 288, 320);
  cvt(bu_pw,  Wbu,  576, 288, 576);
  cvt(sc_w,   Wsc,  576, 288, 576);
  cvt(l10_pw, Wl10, 576, 576, 576);
  cvt(l11_w,  Wl11, 256, 576, 256);
  cvt(l12_w,  Wl12, 576, 256, 576);

  (void)hipMemcpyAsync(Xc, x_in, FULL_E * 4, hipMemcpyDeviceToDevice, stream);

  auto gemm = [&](const float* i1, const float* i2, const _Float16* Wt,
                  const float* g, const float* bt, const float* ad1, const float* ad2,
                  float* o, int Cin, int Cout, int P, int relu) {
    dim3 grid(P / 64, (Cout + 63) / 64, B_);
    gemm_pw_kernel<<<grid, 128, 0, stream>>>(i1, i2, Wt, g, bt, ad1, ad2, o, Cin, Cout, P, relu);
  };
  auto dw = [&](const float* i1, const float* i2, const float* w, float* o, int C, int H, int W) {
    int tot = B_ * C * H * W;
    dw5x5_kernel<<<(tot + 255) / 256, 256, 0, stream>>>(i1, i2, w, o, C, H, W, tot);
  };
  auto mp = [&](const float* i, float* o, int C, int H, int W) {
    int tot = B_ * C * (H / 2) * (W / 2);
    maxpool2_kernel<<<(tot + 255) / 256, 256, 0, stream>>>(i, o, C, H, W, tot);
  };
  auto up = [&](const float* i, float* o, int C, int Hin, int Win) {
    int tot = B_ * C * Hin * 2 * Win * 2;
    upsample2_kernel<<<(tot + 255) / 256, 256, 0, stream>>>(i, o, C, Hin, Win, tot);
  };

  for (int blk = 0; blk < NBLK; ++blk) {
    // ---- reception ----
    dw(Xc, nullptr, a_dw, A1, C_, H_, W_);                                    // A1 = dw(relu(x))
    gemm(A1, nullptr, Wa, a_g, a_b, Xc, nullptr, A2, 576, 576, 1024, 0);      // A2 = a = x + bn(pw)
    mp(Xc, S1, C_, H_, W_);                                                   // S1 = mp(x)
    gemm(S1, nullptr, Wlow, low_g, low_b, nullptr, nullptr, L1, 576, 288, 256, 1); // L1 = low
    dw(L1, nullptr, b1_dw, L2, CHLF, 16, 16);
    gemm(L2, nullptr, Wb1, b1_g, b1_b, L1, nullptr, L3, 288, 288, 256, 0);    // L3 = b
    mp(L1, M1, CHLF, 16, 16);                                                 // M1 = mp(low)
    dw(M1, nullptr, c1_dw, M2, CHLF, 8, 8);
    gemm(M2, nullptr, Wc1, c1_g, c1_b, M1, nullptr, M3, 288, 288, 64, 0);     // M3 = c (c1)
    dw(M3, nullptr, c2_dw, M2, CHLF, 8, 8);
    gemm(M2, nullptr, Wc2, c2_g, c2_b, M3, nullptr, M1, 288, 288, 64, 0);     // M1 = c (c2)
    up(M1, L2, CHLF, 8, 8);                                                   // L2 = up(c)
    dw(L3, L2, bu_dw, L1, CHLF, 16, 16);                                      // L1 = dw(relu(b+c))
    gemm(L1, nullptr, Wbu, bu_g, bu_b, nullptr, nullptr, S1, 288, 576, 256, 0);    // S1 = y
    gemm(L3, L2, Wsc, sc_g, sc_b, S1, nullptr, S2, 288, 576, 256, 1);         // S2 = sc(b+c) + y
    up(S2, A1, C_, 16, 16);                                                   // A1 = up(bup)
    {
      int tot = (int)FULL_E;
      add_kernel<<<(tot + 255) / 256, 256, 0, stream>>>(A2, A1, Xc, tot);     // Xc = x = ident
    }
    // ---- l10 ----
    dw(Xc, nullptr, l10_dw, A1, C_, H_, W_);
    gemm(A1, nullptr, Wl10, l10_g, l10_b, nullptr, nullptr, A2, 576, 576, 1024, 0); // A2 = x
    // ---- l11 + heads ----
    gemm(A2, nullptr, Wl11, nullptr, nullptr, nullptr, nullptr, Hbuf, 576, 256, 1024, 1); // h
    head_stats_kernel<<<B_ * NHEAT, 256, 0, stream>>>(Hbuf, stats);
    head_out_kernel<<<1, 128, 0, stream>>>(stats, outp, blk);
    // ---- l12 residual ----
    if (blk < NBLK - 1)
      gemm(Hbuf, nullptr, Wl12, l12_g, l12_b, Xc, A2, Xc, 256, 576, 1024, 1); // Xc = ident+x+hproj
  }
}
